// LanguageModel_17635135717738
// MI455X (gfx1250) — matrix operand order repaired and checked by reading, not compiled, not run on hardware
//
#include <hip/hip_runtime.h>

// ---- model constants (match reference) ----
#define L_  4
#define H_  8
#define D_  512
#define HD_ 64
#define F_  2048
#define V_  32000
#define B_  2
#define S_  2048
#define M_  (B_ * S_)   // 4096 tokens

typedef __attribute__((ext_vector_type(16))) __bf16 v16bf;
typedef __attribute__((ext_vector_type(8)))  float  v8f;

union FragB {
  v16bf v;
  unsigned int u[8];
  unsigned short h[16];
};

__device__ __forceinline__ unsigned short f2bf(float f) {
  unsigned int u = __float_as_uint(f);
  u += 0x7fffu + ((u >> 16) & 1u);   // round-to-nearest-even
  return (unsigned short)(u >> 16);
}

// A-fragment K packing for 16-bit 16x32: lanes0-15 K=0..15; lanes16-31 K=16..31 (matches B)
__device__ __forceinline__ int apack(int v, int half) {
  return half * 16 + 2 * v;
}

// ---------------- one-shot weight conversion kernels (f32 -> bf16, with re-layout) ----------------
__global__ __launch_bounds__(256) void cvt_plain_kernel(const float* __restrict__ src,
                                                        unsigned short* __restrict__ dst, int n) {
  int i = blockIdx.x * 256 + threadIdx.x;
  if (i < n) dst[i] = f2bf(src[i]);
}

// src [L][H][D][64]  ->  dst [L][D][H*64]  (de-head-block QKV weights)
__global__ __launch_bounds__(256) void cvt_qkv_kernel(const float* __restrict__ src,
                                                      unsigned short* __restrict__ dst) {
  int i = blockIdx.x * 256 + threadIdx.x;  // over L*D*512
  int l = i >> 18;                         // D*512 = 2^18
  int rem = i & 262143;
  int k = rem >> 9;
  int n = rem & 511;
  size_t si = ((size_t)l * H_ + (size_t)(n >> 6)) * ((size_t)D_ * 64) + (size_t)k * 64 + (size_t)(n & 63);
  dst[i] = f2bf(src[si]);
}

// src head_W [V][D]  ->  dst [D][V]  (transpose so logits GEMM is plain [K,N])
__global__ __launch_bounds__(256) void cvt_headT_kernel(const float* __restrict__ src,
                                                        unsigned short* __restrict__ dst) {
  int i = blockIdx.x * 256 + threadIdx.x;  // indexed by source, V*D total
  int n = i >> 9;       // vocab row
  int k = i & 511;      // channel
  dst[(size_t)k * V_ + n] = f2bf(src[i]);
}

// ---------------- embedding + sinusoidal PE ----------------
__global__ __launch_bounds__(256) void embed_kernel(const int* __restrict__ tokens,
                                                    const float* __restrict__ emb,
                                                    float* __restrict__ X,
                                                    unsigned short* __restrict__ Xbf) {
  int idx = blockIdx.x * 256 + threadIdx.x;
  if (idx >= M_ * D_) return;
  int m = idx >> 9;
  int c = idx & (D_ - 1);
  int s = m & (S_ - 1);
  int tok = tokens[m];
  float freq = __expf(-logf(10000.f) * (float)(c & ~1) * (1.f / (float)D_));
  float ang  = (float)s * freq;
  float pe   = (c & 1) ? __cosf(ang) : __sinf(ang);
  float val  = emb[(size_t)tok * D_ + c] + pe;
  X[idx]   = val;
  Xbf[idx] = f2bf(val);
}

// ---------------- layernorm (one wave per 512-wide row) ----------------
__global__ __launch_bounds__(256) void ln_kernel(const float* __restrict__ Y,
                                                 const float* __restrict__ g,
                                                 const float* __restrict__ bta,
                                                 float* __restrict__ Xo,
                                                 unsigned short* __restrict__ Xbf) {
  int wave = threadIdx.x >> 5, lane = threadIdx.x & 31;
  int row = blockIdx.x * 8 + wave;
  const float* yr = Y + (size_t)row * D_;
  float vals[16];
  float s = 0.f, ss = 0.f;
  for (int i = 0; i < 16; ++i) {
    float v = yr[lane + 32 * i];
    vals[i] = v;
    s += v;
    ss += v * v;
  }
  for (int o = 1; o < 32; o <<= 1) {
    s  += __shfl_xor(s, o, 32);
    ss += __shfl_xor(ss, o, 32);
  }
  float mean = s * (1.f / (float)D_);
  float var  = ss * (1.f / (float)D_) - mean * mean;
  float rstd = rsqrtf(var + 1e-5f);
  for (int i = 0; i < 16; ++i) {
    int c = lane + 32 * i;
    float o = (vals[i] - mean) * rstd * g[c] + bta[c];
    Xo[(size_t)row * D_ + c]  = o;
    Xbf[(size_t)row * D_ + c] = f2bf(o);
  }
}

// ---------------- WMMA bf16 GEMM: double-buffered LDS, async A staging ----------------
// C[M,N](f32 acc) = A_bf16[M,K] @ B_bf16[K,N]  (+bias)(+resid)(relu)  -> Cf (f32) and/or Cbf (bf16)
__global__ __launch_bounds__(256) void gemm_bf16_kernel(
    const unsigned short* __restrict__ A, const unsigned short* __restrict__ Bw,
    const float* __restrict__ bias, const float* __restrict__ resid,
    float* __restrict__ Cf, unsigned short* __restrict__ Cbf,
    int M, int N, int K, int relu) {
  __shared__ unsigned short As[2][128][40];  // 80B row stride: rows 16B-aligned for async b128
  __shared__ unsigned short Bs[2][128][36];  // transposed [n][k], u32-aligned fragment reads
  int tid = threadIdx.x;
  int lane = tid & 31, wave = tid >> 5;
  int half = lane >> 4, ln = lane & 15;
  int waveM = wave >> 2, waveN = wave & 3;  // 2x4 wave grid, each wave: 64x32 region
  int m0 = blockIdx.y * 128, n0 = blockIdx.x * 128;

  // stage A tile 128x32 bf16 via async LDS copy (512 x b128, 2 per thread)
  auto stageA = [&](int buf, int k0) {
    for (int t = 0; t < 2; ++t) {
      int i = tid * 2 + t;
      int r = i >> 2, seg = i & 3;  // 4 x 16B segments per 64B row
      unsigned long long ga = (unsigned long long)(const void*)(A + (size_t)(m0 + r) * K + k0 + seg * 8);
      unsigned int la = (unsigned int)(size_t)(void*)&As[buf][r][seg * 8];
      asm volatile("global_load_async_to_lds_b128 %0, %1, off" :: "v"(la), "v"(ga) : "memory");
    }
  };
  // stage B tile 32x128 bf16 transposed: batch loads first, then LDS stores
  auto stageB = [&](int buf, int k0) {
    unsigned int d[8];
    for (int i = 0; i < 8; ++i) {
      int pi = tid + (i << 8);
      int k = pi >> 6, np = pi & 63;
      d[i] = *(const unsigned int*)(Bw + (size_t)(k0 + k) * N + n0 + 2 * np);
    }
    for (int i = 0; i < 8; ++i) {
      int pi = tid + (i << 8);
      int k = pi >> 6, np = pi & 63;
      Bs[buf][2 * np][k]     = (unsigned short)(d[i] & 0xffffu);
      Bs[buf][2 * np + 1][k] = (unsigned short)(d[i] >> 16);
    }
  };

  v8f zero = {};
  v8f acc[4][2];
  for (int i = 0; i < 4; ++i)
    for (int j = 0; j < 2; ++j) acc[i][j] = zero;

  stageA(0, 0);
  stageB(0, 0);
  asm volatile("s_wait_asynccnt 0" ::: "memory");
  __syncthreads();

  int nk = K >> 5;
  for (int it = 0; it < nk; ++it) {
    int cur = it & 1, nxt = cur ^ 1;
    // overlap: kick off next tile (async A + batched B) before computing this one
    if (it + 1 < nk) {
      int k0n = (it + 1) << 5;
      stageA(nxt, k0n);
      stageB(nxt, k0n);
    }
    if (it + 2 < nk) {  // pull tile after next into GL2
      int k0p = (it + 2) << 5;
      __builtin_prefetch(A + (size_t)(m0 + (tid >> 1)) * K + k0p + (tid & 1) * 16, 0, 1);
      __builtin_prefetch(Bw + (size_t)(k0p + (tid >> 3)) * N + n0 + (tid & 7) * 16, 0, 1);
    }

    FragB a[4], b[2];
    for (int i = 0; i < 4; ++i) {
      int r = waveM * 64 + 16 * i + ln;
      for (int v = 0; v < 8; ++v)
        a[i].u[v] = *(const unsigned int*)&As[cur][r][apack(v, half)];
    }
    for (int j = 0; j < 2; ++j) {
      int n = waveN * 32 + 16 * j + ln;
      for (int v = 0; v < 8; ++v)
        b[j].u[v] = *(const unsigned int*)&Bs[cur][n][half * 16 + 2 * v];
    }
    for (int i = 0; i < 4; ++i)
      for (int j = 0; j < 2; ++j)
        acc[i][j] = __builtin_amdgcn_wmma_f32_16x16x32_bf16(
            false, a[i].v, false, b[j].v, (short)0, acc[i][j], false, false);

    asm volatile("s_wait_asynccnt 0" ::: "memory");
    __syncthreads();
  }

  // epilogue: C layout = lanes0-15: (M=p, N=ln); lanes16-31: (M=8+p, N=ln)
  for (int i = 0; i < 4; ++i) {
    for (int j = 0; j < 2; ++j) {
      int tm = m0 + waveM * 64 + 16 * i;
      int tn = n0 + waveN * 32 + 16 * j;
      for (int p = 0; p < 8; ++p) {
        int mg = tm + p + 8 * half;
        int ng = tn + ln;
        float val = acc[i][j][p];
        if (bias)  val += bias[ng];
        if (resid) val += resid[(size_t)mg * N + ng];
        if (relu)  val = fmaxf(val, 0.f);
        if (Cf)  Cf[(size_t)mg * N + ng]  = val;
        if (Cbf) Cbf[(size_t)mg * N + ng] = f2bf(val);
      }
    }
  }
}

// ---------------- flash attention (causal, per-head) ----------------
// Q,K,V,O: bf16 [B,S,D] with head h occupying cols h*64..h*64+63.
// Block = 8 waves x 16 query rows = 128 queries. Each wave independent (no workgroup barriers).
__global__ __launch_bounds__(256) void attn_kernel(const unsigned short* __restrict__ Q,
                                                   const unsigned short* __restrict__ Kb,
                                                   const unsigned short* __restrict__ Vb,
                                                   unsigned short* __restrict__ Ob) {
  __shared__ unsigned short Pl[8][512];   // per-wave P tile 16x32 bf16
  __shared__ unsigned short Vt[8][2048];  // per-wave V tile transposed [e(64)][t(32)]
  int lane = threadIdx.x & 31, wave = threadIdx.x >> 5;
  int half = lane >> 4, ln = lane & 15;
  int bh = blockIdx.y;
  int b = bh / H_, h = bh % H_;
  int qrow0 = blockIdx.x * 128 + wave * 16;
  const size_t hoff = (size_t)h * HD_;

  FragB qa0, qa1;
  {
    const unsigned short* qp = Q + ((size_t)(b * S_ + qrow0 + ln)) * D_ + hoff;
    for (int v = 0; v < 8; ++v) {
      int k = apack(v, half);
      qa0.u[v] = *(const unsigned int*)(qp + k);
      qa1.u[v] = *(const unsigned int*)(qp + 32 + k);
    }
  }

  v8f zero = {};
  v8f o0 = zero, o1 = zero, o2 = zero, o3 = zero;
  float mrow[8], lrow[8];
  for (int p = 0; p < 8; ++p) { mrow[p] = -1e30f; lrow[p] = 0.f; }
  const float scale = 0.125f;  // 64^-0.5

  for (int t0 = 0; t0 <= qrow0 + 15; t0 += 32) {
    FragB kb00, kb01, kb10, kb11;
    {
      const unsigned short* kp0 = Kb + ((size_t)(b * S_ + t0 + ln)) * D_ + hoff;
      const unsigned short* kp1 = kp0 + (size_t)16 * D_;
      for (int v = 0; v < 8; ++v) {
        int e = half * 16 + 2 * v;
        kb00.u[v] = *(const unsigned int*)(kp0 + e);
        kb01.u[v] = *(const unsigned int*)(kp0 + 32 + e);
        kb10.u[v] = *(const unsigned int*)(kp1 + e);
        kb11.u[v] = *(const unsigned int*)(kp1 + 32 + e);
      }
    }
    v8f s0 = zero, s1 = zero;
    s0 = __builtin_amdgcn_wmma_f32_16x16x32_bf16(false, qa0.v, false, kb00.v, (short)0, s0, false, false);
    s0 = __builtin_amdgcn_wmma_f32_16x16x32_bf16(false, qa1.v, false, kb01.v, (short)0, s0, false, false);
    s1 = __builtin_amdgcn_wmma_f32_16x16x32_bf16(false, qa0.v, false, kb10.v, (short)0, s1, false, false);
    s1 = __builtin_amdgcn_wmma_f32_16x16x32_bf16(false, qa1.v, false, kb11.v, (short)0, s1, false, false);

    // stage V block [t:32][e:64] transposed into per-wave LDS, batched (8 loads then 16 stores)
    for (int pass = 0; pass < 4; ++pass) {
      unsigned int d[8];
      for (int i = 0; i < 8; ++i) {
        int flat = lane + 32 * (pass * 8 + i);
        int t = flat >> 5, ep = flat & 31;
        d[i] = *(const unsigned int*)(Vb + ((size_t)(b * S_ + t0 + t)) * D_ + hoff + 2 * ep);
      }
      for (int i = 0; i < 8; ++i) {
        int flat = lane + 32 * (pass * 8 + i);
        int t = flat >> 5, ep = flat & 31;
        Vt[wave][(2 * ep) * 32 + t]     = (unsigned short)(d[i] & 0xffffu);
        Vt[wave][(2 * ep + 1) * 32 + t] = (unsigned short)(d[i] >> 16);
      }
    }

    // online softmax; row p+8*half lives in VGPR p across the lane half
    for (int p = 0; p < 8; ++p) {
      int row = qrow0 + p + 8 * half;
      int c0 = t0 + ln;
      float a0 = (c0 <= row) ? s0[p] * scale : -1e30f;
      float a1 = (c0 + 16 <= row) ? s1[p] * scale : -1e30f;
      float rmax = fmaxf(a0, a1);
      for (int o = 1; o < 16; o <<= 1) rmax = fmaxf(rmax, __shfl_xor(rmax, o, 16));
      float mnew = fmaxf(mrow[p], rmax);
      float alpha = __expf(mrow[p] - mnew);
      float p0 = __expf(a0 - mnew);
      float p1 = __expf(a1 - mnew);
      float rsum = p0 + p1;
      for (int o = 1; o < 16; o <<= 1) rsum += __shfl_xor(rsum, o, 16);
      lrow[p] = lrow[p] * alpha + rsum;
      mrow[p] = mnew;
      o0[p] *= alpha; o1[p] *= alpha; o2[p] *= alpha; o3[p] *= alpha;
      int ri = p + 8 * half;
      Pl[wave][ri * 32 + ln]      = f2bf(p0);
      Pl[wave][ri * 32 + 16 + ln] = f2bf(p1);
    }
    asm volatile("s_wait_dscnt 0" ::: "memory");  // wave-local LDS RAW fence

    FragB pa;
    for (int v = 0; v < 8; ++v)
      pa.u[v] = *(const unsigned int*)&Pl[wave][ln * 32 + apack(v, half)];
    FragB vb0, vb1, vb2, vb3;
    for (int v = 0; v < 8; ++v) {
      int kk = half * 16 + 2 * v;
      vb0.u[v] = *(const unsigned int*)&Vt[wave][(0 * 16 + ln) * 32 + kk];
      vb1.u[v] = *(const unsigned int*)&Vt[wave][(1 * 16 + ln) * 32 + kk];
      vb2.u[v] = *(const unsigned int*)&Vt[wave][(2 * 16 + ln) * 32 + kk];
      vb3.u[v] = *(const unsigned int*)&Vt[wave][(3 * 16 + ln) * 32 + kk];
    }
    o0 = __builtin_amdgcn_wmma_f32_16x16x32_bf16(false, pa.v, false, vb0.v, (short)0, o0, false, false);
    o1 = __builtin_amdgcn_wmma_f32_16x16x32_bf16(false, pa.v, false, vb1.v, (short)0, o1, false, false);
    o2 = __builtin_amdgcn_wmma_f32_16x16x32_bf16(false, pa.v, false, vb2.v, (short)0, o2, false, false);
    o3 = __builtin_amdgcn_wmma_f32_16x16x32_bf16(false, pa.v, false, vb3.v, (short)0, o3, false, false);
  }

  for (int p = 0; p < 8; ++p) {
    int row = qrow0 + p + 8 * half;
    float inv = 1.f / lrow[p];
    unsigned short* op = Ob + ((size_t)(b * S_ + row)) * D_ + hoff;
    op[0 * 16 + ln] = f2bf(o0[p] * inv);
    op[1 * 16 + ln] = f2bf(o1[p] * inv);
    op[2 * 16 + ln] = f2bf(o2[p] * inv);
    op[3 * 16 + ln] = f2bf(o3[p] * inv);
  }
}

// ---------------- host orchestration ----------------
extern "C" void kernel_launch(void* const* d_in, const int* in_sizes, int n_in,
                              void* d_out, int out_size, void* d_ws, size_t ws_size,
                              hipStream_t stream) {
  (void)in_sizes; (void)n_in; (void)out_size; (void)ws_size;
  const int*   tokens = (const int*)d_in[0];
  const float* emb    = (const float*)d_in[1];
  const float* Wq     = (const float*)d_in[2];
  const float* Wk     = (const float*)d_in[3];
  const float* Wv     = (const float*)d_in[4];
  const float* Wo     = (const float*)d_in[5];
  const float* bo     = (const float*)d_in[6];
  const float* ln1g   = (const float*)d_in[7];
  const float* ln1b   = (const float*)d_in[8];
  const float* W1     = (const float*)d_in[9];
  const float* b1     = (const float*)d_in[10];
  const float* W2     = (const float*)d_in[11];
  const float* b2     = (const float*)d_in[12];
  const float* ln2g   = (const float*)d_in[13];
  const float* ln2b   = (const float*)d_in[14];
  const float* lnfg   = (const float*)d_in[15];
  const float* lnfb   = (const float*)d_in[16];
  const float* headW  = (const float*)d_in[17];
  float* out = (float*)d_out;

  char* w = (char*)d_ws;
  float*          X    = (float*)(w + 0);                  //  8 MB  [M,D] f32
  float*          Y    = (float*)(w + 8388608);            //  8 MB  [M,D] f32 (pre-LN)
  unsigned short* Xbf  = (unsigned short*)(w + 16777216);  //  4 MB  [M,D] bf16
  unsigned short* Qb   = (unsigned short*)(w + 20971520);  //  4 MB
  unsigned short* Kbuf = (unsigned short*)(w + 25165824);  //  4 MB
  unsigned short* Vbuf = (unsigned short*)(w + 29360128);  //  4 MB
  unsigned short* Ab   = (unsigned short*)(w + 33554432);  //  4 MB  attn out
  unsigned short* Hb   = (unsigned short*)(w + 37748736);  // 16 MB  [M,F] bf16
  unsigned short* Wqb  = (unsigned short*)(w + 54525952);  //  2 MB  [L][D][512] bf16
  unsigned short* Wkb  = (unsigned short*)(w + 56623104);  //  2 MB
  unsigned short* Wvb  = (unsigned short*)(w + 58720256);  //  2 MB
  unsigned short* Wob  = (unsigned short*)(w + 60817408);  //  2 MB  [L][D][D]
  unsigned short* W1b  = (unsigned short*)(w + 62914560);  //  8 MB  [L][D][F]
  unsigned short* W2b  = (unsigned short*)(w + 71303168);  //  8 MB  [L][F][D]
  unsigned short* HTb  = (unsigned short*)(w + 79691776);  // 32 MB  [D][V]

  // ---- weight preconversion (bf16, WMMA-ready layouts) ----
  cvt_qkv_kernel<<<(L_ * D_ * 512) / 256, 256, 0, stream>>>(Wq, Wqb);
  cvt_qkv_kernel<<<(L_ * D_ * 512) / 256, 256, 0, stream>>>(Wk, Wkb);
  cvt_qkv_kernel<<<(L_ * D_ * 512) / 256, 256, 0, stream>>>(Wv, Wvb);
  cvt_plain_kernel<<<(L_ * D_ * D_) / 256, 256, 0, stream>>>(Wo, Wob, L_ * D_ * D_);
  cvt_plain_kernel<<<(L_ * D_ * F_) / 256, 256, 0, stream>>>(W1, W1b, L_ * D_ * F_);
  cvt_plain_kernel<<<(L_ * F_ * D_) / 256, 256, 0, stream>>>(W2, W2b, L_ * F_ * D_);
  cvt_headT_kernel<<<(V_ * D_) / 256, 256, 0, stream>>>(headW, HTb);

  embed_kernel<<<(M_ * D_) / 256, 256, 0, stream>>>(tokens, emb, X, Xbf);

  for (int l = 0; l < L_; ++l) {
    gemm_bf16_kernel<<<dim3(D_ / 128, M_ / 128), 256, 0, stream>>>(
        Xbf, Wqb + (size_t)l * D_ * 512, nullptr, nullptr, nullptr, Qb, M_, D_, D_, 0);
    gemm_bf16_kernel<<<dim3(D_ / 128, M_ / 128), 256, 0, stream>>>(
        Xbf, Wkb + (size_t)l * D_ * 512, nullptr, nullptr, nullptr, Kbuf, M_, D_, D_, 0);
    gemm_bf16_kernel<<<dim3(D_ / 128, M_ / 128), 256, 0, stream>>>(
        Xbf, Wvb + (size_t)l * D_ * 512, nullptr, nullptr, nullptr, Vbuf, M_, D_, D_, 0);
    attn_kernel<<<dim3(S_ / 128, B_ * H_), 256, 0, stream>>>(Qb, Kbuf, Vbuf, Ab);
    gemm_bf16_kernel<<<dim3(D_ / 128, M_ / 128), 256, 0, stream>>>(
        Ab, Wob + (size_t)l * D_ * D_, bo + (size_t)l * D_, X, Y, nullptr, M_, D_, D_, 0);
    ln_kernel<<<M_ / 8, 256, 0, stream>>>(Y, ln1g + (size_t)l * D_, ln1b + (size_t)l * D_, X, Xbf);
    gemm_bf16_kernel<<<dim3(F_ / 128, M_ / 128), 256, 0, stream>>>(
        Xbf, W1b + (size_t)l * D_ * F_, b1 + (size_t)l * F_, nullptr, nullptr, Hb, M_, F_, D_, 1);
    gemm_bf16_kernel<<<dim3(D_ / 128, M_ / 128), 256, 0, stream>>>(
        Hb, W2b + (size_t)l * F_ * D_, b2 + (size_t)l * D_, X, Y, nullptr, M_, D_, F_, 0);
    ln_kernel<<<M_ / 8, 256, 0, stream>>>(Y, ln2g + (size_t)l * D_, ln2b + (size_t)l * D_, X, Xbf);
  }

  ln_kernel<<<M_ / 8, 256, 0, stream>>>(X, lnfg, lnfb, Y, Xbf);
  gemm_bf16_kernel<<<dim3(V_ / 128, M_ / 128), 256, 0, stream>>>(
      Xbf, HTb, nullptr, nullptr, out, nullptr, M_, V_, D_, 0);
}
